// MSA_51075751084037
// MI455X (gfx1250) — compile-verified, matrix operand order checked by reading, not run on hardware
//
#include <hip/hip_runtime.h>

typedef __attribute__((ext_vector_type(16))) _Float16 v16h;
typedef __attribute__((ext_vector_type(8)))  float    v8f;

#define D_MODEL 1024
#define NHEAD   16
#define DK      64
#define SEQ     2048
#define BATCH   2
#define M_TOT   (BATCH*SEQ)   // 4096

union FragU { v16h v; uint4 q[2]; };

// 16-lane-striped 16x32 f16 fragment (A or B operand) from row-major source.
// base is a uniform (SGPR) pointer; off is a per-lane 32-bit element offset,
// so loads lower to global_load_b128 saddr+voffset with immediate ioffset.
// CDNA5 layout: VGPR0-3 hold K = hi*8 + 0..7, VGPR4-7 hold K = 16 + hi*8 + 0..7.
__device__ __forceinline__ v16h load_frag(const _Float16* base, unsigned off, unsigned hi8) {
  FragU u;
  u.q[0] = *reinterpret_cast<const uint4*>(base + off + hi8);
  u.q[1] = *reinterpret_cast<const uint4*>(base + off + hi8 + 16);
  return u.v;
}

#define WMMA(A, B, C) __builtin_amdgcn_wmma_f32_16x16x32_f16(false, (A), false, (B), (short)0, (C), false, false)

__global__ void k_cvt(const float* __restrict__ src, _Float16* __restrict__ dst, int n) {
  int i = blockIdx.x * blockDim.x + threadIdx.x;
  int stride = gridDim.x * blockDim.x;
  for (; i < n; i += stride) dst[i] = (_Float16)src[i];
}

// Fused Q/K/V projection: one wave per 32x32 output tile (2 A frags + 6 B frags
// feed 12 WMMAs per k-step), 2-stage register pipeline.
// Q,K stored [bh][n][64]; V stored transposed [bh][64][n].
__global__ void __launch_bounds__(256)
k_qkv(const _Float16* __restrict__ xh,
      const _Float16* __restrict__ wq, const _Float16* __restrict__ wk,
      const _Float16* __restrict__ wv,
      const float* __restrict__ bq, const float* __restrict__ bk,
      const float* __restrict__ bv,
      _Float16* __restrict__ Qh, _Float16* __restrict__ Kh,
      _Float16* __restrict__ Vt) {
  int wave = (blockIdx.x * blockDim.x + threadIdx.x) >> 5;
  int lane = threadIdx.x & 31;
  int lo = lane & 15, hi = lane >> 4;
  unsigned hi8 = (unsigned)(hi * 8);
  int mt = wave >> 5;               // 0..127  (32-row strips)
  int nt = wave & 31;               // 0..31   (32-col strips)
  int m0 = mt * 32, n0 = nt * 32;

  unsigned ao0 = (unsigned)(m0 + lo) * D_MODEL;
  unsigned ao1 = ao0 + 16u * D_MODEL;
  unsigned bo0 = (unsigned)(n0 + lo) * D_MODEL;
  unsigned bo1 = bo0 + 16u * D_MODEL;

  v8f acc[3][2][2] = {};

  auto load_tiles = [&](int ks, v16h A[2], v16h B[3][2]) {
    A[0]    = load_frag(xh, ao0 + ks, hi8);
    A[1]    = load_frag(xh, ao1 + ks, hi8);
    B[0][0] = load_frag(wq, bo0 + ks, hi8);
    B[0][1] = load_frag(wq, bo1 + ks, hi8);
    B[1][0] = load_frag(wk, bo0 + ks, hi8);
    B[1][1] = load_frag(wk, bo1 + ks, hi8);
    B[2][0] = load_frag(wv, bo0 + ks, hi8);
    B[2][1] = load_frag(wv, bo1 + ks, hi8);
  };
  auto do_wmma = [&](v16h A[2], v16h B[3][2]) {
#pragma unroll
    for (int w = 0; w < 3; ++w)
#pragma unroll
      for (int i = 0; i < 2; ++i)
#pragma unroll
        for (int j = 0; j < 2; ++j)
          acc[w][i][j] = WMMA(A[i], B[w][j], acc[w][i][j]);
  };

  v16h a[2], b[3][2];
  load_tiles(0, a, b);
#pragma unroll 2
  for (int ks = 32; ks < D_MODEL; ks += 32) {
    v16h na[2], nb[3][2];
    load_tiles(ks, na, nb);          // next stage in flight during WMMAs
    do_wmma(a, b);
#pragma unroll
    for (int i = 0; i < 2; ++i) a[i] = na[i];
#pragma unroll
    for (int w = 0; w < 3; ++w)
#pragma unroll
      for (int j = 0; j < 2; ++j) b[w][j] = nb[w][j];
  }
  do_wmma(a, b);

#pragma unroll
  for (int j = 0; j < 2; ++j) {
    int col = n0 + j * 16 + lo;           // output feature 0..1023
    int h = col >> 6, dk = col & 63;
    float vbq = bq[col], vbk = bk[col], vbv = bv[col];
#pragma unroll
    for (int i = 0; i < 2; ++i) {
#pragma unroll
      for (int e = 0; e < 8; ++e) {
        int m = m0 + i * 16 + e + 8 * hi;
        int bb = m >> 11, n = m & (SEQ - 1);
        int bh = bb * NHEAD + h;
        Qh[((size_t)bh * SEQ + n) * DK + dk] = (_Float16)(acc[0][i][j][e] + vbq);
        Kh[((size_t)bh * SEQ + n) * DK + dk] = (_Float16)(acc[1][i][j][e] + vbk);
        Vt[((size_t)bh * DK + dk) * SEQ + n] = (_Float16)(acc[2][i][j][e] + vbv);
      }
    }
  }
}

// Flash attention: one wave per (bh, 16-query tile); streams 32 keys/step.
// K/V fragments for step k+1 are issued right after step k's score WMMAs so
// the softmax VALU chain hides their latency; prefetch 2 steps ahead.
__global__ void __launch_bounds__(256)
k_attn(const _Float16* __restrict__ Qh, const _Float16* __restrict__ Kh,
       const _Float16* __restrict__ Vt, _Float16* __restrict__ ATT) {
  __shared__ _Float16 pbuf[8][16 * 32];   // per-wave P staging (C-layout -> A-layout)
  int wid = threadIdx.x >> 5;
  int lane = threadIdx.x & 31;
  int lo = lane & 15, hi = lane >> 4;
  unsigned hi8 = (unsigned)(hi * 8);
  int wave = blockIdx.x * 8 + wid;
  int bh = wave >> 7;                  // 0..31
  int qt = wave & 127;                 // 0..127
  int q0 = qt * 16;

  const _Float16* Qb = Qh + (size_t)bh * SEQ * DK;
  const _Float16* Kb = Kh + (size_t)bh * SEQ * DK;
  const _Float16* Vb = Vt + (size_t)bh * DK * SEQ;

  unsigned qoff = (unsigned)(q0 + lo) * DK;
  v16h aq0 = load_frag(Qb, qoff, hi8);        // dk 0..31
  v16h aq1 = load_frag(Qb, qoff + 32, hi8);   // dk 32..63

  auto load_k = [&](int key0, v16h kf[4]) {
    unsigned r0 = (unsigned)(key0 + lo) * DK;
    unsigned r1 = (unsigned)(key0 + 16 + lo) * DK;
    kf[0] = load_frag(Kb, r0, hi8);
    kf[1] = load_frag(Kb, r0 + 32, hi8);
    kf[2] = load_frag(Kb, r1, hi8);
    kf[3] = load_frag(Kb, r1 + 32, hi8);
  };
  auto load_v = [&](int key0, v16h vf[4]) {
#pragma unroll
    for (int j = 0; j < 4; ++j)
      vf[j] = load_frag(Vb, (unsigned)(j * 16 + lo) * SEQ + key0, hi8);
  };

  const float sc = 0.125f * 1.44269504088896340736f;  // 1/sqrt(64) * log2(e)
  float mrow[8], lrow[8];
#pragma unroll
  for (int e = 0; e < 8; ++e) { mrow[e] = -1e30f; lrow[e] = 0.f; }
  v8f o[4] = {{}, {}, {}, {}};
  _Float16* pb = pbuf[wid];

  v16h kf[4], vf[4];
  load_k(0, kf);
  load_v(0, vf);

  for (int key0 = 0; key0 < SEQ; key0 += 32) {
    // scores: two 16-key groups, dk = 64 -> 2 k-steps each
    v8f s0 = {}, s1 = {};
    s0 = WMMA(aq0, kf[0], s0);
    s0 = WMMA(aq1, kf[1], s0);
    s1 = WMMA(aq0, kf[2], s1);
    s1 = WMMA(aq1, kf[3], s1);

    bool more = (key0 + 32) < SEQ;
    v16h kf2[4], vf2[4];
    if (more) {                       // issue next-step loads before softmax
      load_k(key0 + 32, kf2);
      load_v(key0 + 32, vf2);
    }
    if (key0 + 64 < SEQ) {            // global_prefetch two steps ahead
      __builtin_prefetch(Kb + (size_t)(key0 + 64 + lo) * DK, 0, 3);
      __builtin_prefetch(Vb + (size_t)lo * SEQ + key0 + 64, 0, 3);
    }

    // online softmax in exp2 domain; row e+8*hi lives across each 16-lane half
    float sf[8];
#pragma unroll
    for (int e = 0; e < 8; ++e) {
      float a0 = s0[e] * sc, a1 = s1[e] * sc;
      float t = fmaxf(a0, a1);
#pragma unroll
      for (int msk = 1; msk < 16; msk <<= 1) t = fmaxf(t, __shfl_xor(t, msk, 32));
      float nm = fmaxf(mrow[e], t);
      sf[e] = exp2f(mrow[e] - nm);
      float p0 = exp2f(a0 - nm);
      float p1 = exp2f(a1 - nm);
      s0[e] = p0; s1[e] = p1;
      float r = p0 + p1;
#pragma unroll
      for (int msk = 1; msk < 16; msk <<= 1) r += __shfl_xor(r, msk, 32);
      lrow[e] = lrow[e] * sf[e] + r;
      mrow[e] = nm;
    }
    // rescale O, stage P to LDS row-major [16 q][32 key]
#pragma unroll
    for (int e = 0; e < 8; ++e) {
      o[0][e] *= sf[e]; o[1][e] *= sf[e]; o[2][e] *= sf[e]; o[3][e] *= sf[e];
      int row = e + 8 * hi;
      pb[row * 32 + lo]      = (_Float16)s0[e];
      pb[row * 32 + 16 + lo] = (_Float16)s1[e];
    }
    // reload P as A fragment (same wave: DS ops in order, DScnt waited by compiler)
    v16h pf = load_frag(pb, (unsigned)(lo * 32), hi8);
    // O += P * V  (V transposed: [d][key], contiguous in key)
#pragma unroll
    for (int j = 0; j < 4; ++j)
      o[j] = WMMA(pf, vf[j], o[j]);

    if (more) {
#pragma unroll
      for (int j = 0; j < 4; ++j) { kf[j] = kf2[j]; vf[j] = vf2[j]; }
    }
  }

  int b = bh >> 4, h = bh & 15;
  float inv[8];
#pragma unroll
  for (int e = 0; e < 8; ++e) inv[e] = 1.0f / lrow[e];
#pragma unroll
  for (int j = 0; j < 4; ++j) {
    int d = h * DK + j * 16 + lo;
#pragma unroll
    for (int e = 0; e < 8; ++e) {
      int n = q0 + e + 8 * hi;
      ATT[((size_t)(b * SEQ + n)) * D_MODEL + d] = (_Float16)(o[j][e] * inv[e]);
    }
  }
}

// OUT = ATT @ Wo^T + bo (fp32 output). One wave per 32x64 tile, pipelined.
__global__ void __launch_bounds__(256)
k_out(const _Float16* __restrict__ ATT, const _Float16* __restrict__ wo,
      const float* __restrict__ bo, float* __restrict__ out) {
  int wave = (blockIdx.x * blockDim.x + threadIdx.x) >> 5;
  int lane = threadIdx.x & 31;
  int lo = lane & 15, hi = lane >> 4;
  unsigned hi8 = (unsigned)(hi * 8);
  int mt = wave >> 4;        // 0..127 (32-row strips)
  int ng = wave & 15;        // 0..15  (64 cols)
  int m0 = mt * 32, n0 = ng * 64;

  unsigned ao0 = (unsigned)(m0 + lo) * D_MODEL;
  unsigned ao1 = ao0 + 16u * D_MODEL;
  unsigned bo0 = (unsigned)(n0 + lo) * D_MODEL;

  v8f acc[2][4] = {};

  auto load_tiles = [&](int ks, v16h A[2], v16h B[4]) {
    A[0] = load_frag(ATT, ao0 + ks, hi8);
    A[1] = load_frag(ATT, ao1 + ks, hi8);
#pragma unroll
    for (int j = 0; j < 4; ++j)
      B[j] = load_frag(wo, bo0 + (unsigned)j * 16u * D_MODEL + ks, hi8);
  };
  auto do_wmma = [&](v16h A[2], v16h B[4]) {
#pragma unroll
    for (int i = 0; i < 2; ++i)
#pragma unroll
      for (int j = 0; j < 4; ++j)
        acc[i][j] = WMMA(A[i], B[j], acc[i][j]);
  };

  v16h a[2], b[4];
  load_tiles(0, a, b);
#pragma unroll 2
  for (int ks = 32; ks < D_MODEL; ks += 32) {
    v16h na[2], nb[4];
    load_tiles(ks, na, nb);
    do_wmma(a, b);
#pragma unroll
    for (int i = 0; i < 2; ++i) a[i] = na[i];
#pragma unroll
    for (int j = 0; j < 4; ++j) b[j] = nb[j];
  }
  do_wmma(a, b);

#pragma unroll
  for (int j = 0; j < 4; ++j) {
    int col = n0 + j * 16 + lo;
    float bb = bo[col];
#pragma unroll
    for (int i = 0; i < 2; ++i) {
#pragma unroll
      for (int e = 0; e < 8; ++e) {
        int m = m0 + i * 16 + e + 8 * hi;
        out[(size_t)m * D_MODEL + col] = acc[i][j][e] + bb;
      }
    }
  }
}

extern "C" void kernel_launch(void* const* d_in, const int* in_sizes, int n_in,
                              void* d_out, int out_size, void* d_ws, size_t ws_size,
                              hipStream_t stream) {
  const float* x  = (const float*)d_in[0];
  const float* Wq = (const float*)d_in[1];
  const float* bq = (const float*)d_in[2];
  const float* Wk = (const float*)d_in[3];
  const float* bk = (const float*)d_in[4];
  const float* Wv = (const float*)d_in[5];
  const float* bv = (const float*)d_in[6];
  const float* Wo = (const float*)d_in[7];
  const float* bo = (const float*)d_in[8];
  float* out = (float*)d_out;

  // workspace layout (f16 elements): 48 MiB total
  _Float16* p   = (_Float16*)d_ws;
  _Float16* xh  = p;  p += (size_t)M_TOT * D_MODEL;        // 4096x1024
  _Float16* wqh = p;  p += (size_t)D_MODEL * D_MODEL;
  _Float16* wkh = p;  p += (size_t)D_MODEL * D_MODEL;
  _Float16* wvh = p;  p += (size_t)D_MODEL * D_MODEL;
  _Float16* woh = p;  p += (size_t)D_MODEL * D_MODEL;
  _Float16* Qh  = p;  p += (size_t)BATCH * NHEAD * SEQ * DK;
  _Float16* Kh  = p;  p += (size_t)BATCH * NHEAD * SEQ * DK;
  _Float16* Vt  = p;  p += (size_t)BATCH * NHEAD * SEQ * DK;
  _Float16* ATT = p;  p += (size_t)M_TOT * D_MODEL;

  k_cvt<<<1024, 256, 0, stream>>>(x,  xh,  M_TOT * D_MODEL);
  k_cvt<<<256,  256, 0, stream>>>(Wq, wqh, D_MODEL * D_MODEL);
  k_cvt<<<256,  256, 0, stream>>>(Wk, wkh, D_MODEL * D_MODEL);
  k_cvt<<<256,  256, 0, stream>>>(Wv, wvh, D_MODEL * D_MODEL);
  k_cvt<<<256,  256, 0, stream>>>(Wo, woh, D_MODEL * D_MODEL);

  // 128 m-strips * 32 n-strips = 4096 waves / 8 per block
  k_qkv<<<512, 256, 0, stream>>>(xh, wqh, wkh, wvh, bq, bk, bv, Qh, Kh, Vt);

  // 32 bh * 128 q-tiles = 4096 waves / 8 per block
  k_attn<<<512, 256, 0, stream>>>(Qh, Kh, Vt, ATT);

  // 128 m-strips * 16 n-strips = 2048 waves / 8 per block
  k_out<<<256, 256, 0, stream>>>(ATT, woh, bo, out);
}
